// SNMA_81990925681088
// MI455X (gfx1250) — matrix operand order repaired and checked by reading, not compiled, not run on hardware
//
#include <hip/hip_runtime.h>
#include <hip/hip_bf16.h>
#include <math.h>

// dims
#define B_   128
#define S_   64
#define FIN_ 64
#define M_   256
#define L_   24
#define KTOT 16384   // RANK*FEAT = 16*1024
#define HD   64      // 2*BD

typedef __attribute__((ext_vector_type(16))) __bf16 v16bf;
typedef __attribute__((ext_vector_type(8)))  float  v8f;

__device__ __forceinline__ float gelu_exact(float x){
    return 0.5f * x * (1.0f + erff(x * 0.70710678118654752f));
}

// branchless tanh: 1 - 2/(exp2(2*log2e*x)+1); hw v_exp_f32 + v_rcp_f32,
// saturates correctly for |x|->inf (exp2->inf => rcp->0 => +1; exp2->0 => -1)
__device__ __forceinline__ float tanh_fast(float x){
    float e = __builtin_amdgcn_exp2f(x * 2.8853900817779268f);
    return 1.0f - 2.0f * __builtin_amdgcn_rcpf(e + 1.0f);
}

// round-to-nearest-even f32 -> bf16
__device__ __forceinline__ __bf16 f2bf(float f){
    union { float f; unsigned u; } v; v.f = f;
    unsigned r = v.u + 0x7FFFu + ((v.u >> 16) & 1u);
    unsigned short h = (unsigned short)(r >> 16);
    __bf16 b; __builtin_memcpy(&b, &h, 2);
    return b;
}

// 256-thread (8-wave, wave32) block sum
__device__ __forceinline__ float blk_sum(float v, float* s){
    #pragma unroll
    for (int o = 16; o > 0; o >>= 1) v += __shfl_xor(v, o, 32);
    int lane = threadIdx.x & 31, wid = threadIdx.x >> 5;
    __syncthreads();
    if (lane == 0) s[wid] = v;
    __syncthreads();
    if (threadIdx.x == 0){
        float t = 0.f;
        #pragma unroll
        for (int i = 0; i < 8; i++) t += s[i];
        s[0] = t;
    }
    __syncthreads();
    return s[0];
}

// ---------------- Stage 1: encoder + token LN + GELU + mean over S ----------------
__global__ void __launch_bounds__(256) k_encoder(
        const float* __restrict__ pogt, const float* __restrict__ We,
        const float* __restrict__ be,   const float* __restrict__ lng,
        const float* __restrict__ lnb,  float* __restrict__ enc){
    const int b = blockIdx.x, j = threadIdx.x;
    __shared__ float xrow[FIN_];
    __shared__ float red[8];
    float wcol[FIN_];
    #pragma unroll
    for (int d = 0; d < FIN_; d++) wcol[d] = We[d * M_ + j];
    const float bej = be[j], gj = lng[j], bj = lnb[j];
    float acc = 0.f;
    for (int s = 0; s < S_; s++){
        if (j < FIN_) xrow[j] = pogt[(b * S_ + s) * FIN_ + j];
        __syncthreads();
        float y = bej;
        #pragma unroll
        for (int d = 0; d < FIN_; d++) y = fmaf(xrow[d], wcol[d], y);
        float sm = blk_sum(y, red);
        float sq = blk_sum(y * y, red);
        float mu  = sm * (1.f / M_);
        float var = sq * (1.f / M_) - mu * mu;
        float xn  = (y - mu) * rsqrtf(var + 1e-5f) * gj + bj;
        acc += gelu_exact(xn);
    }
    enc[b * M_ + j] = acc * (1.f / S_);
}

// ---------- Stage 2: predictor, surprise, gate (memory==0), mem_new, hypernet enc ----------
__global__ void __launch_bounds__(256) k_state(
        const float* __restrict__ enc,
        const float* __restrict__ Wp1, const float* __restrict__ bp1,
        const float* __restrict__ Wp2, const float* __restrict__ bp2,
        const float* __restrict__ bg1,
        const float* __restrict__ lngg, const float* __restrict__ lngb,
        const float* __restrict__ Wg2,  const float* __restrict__ bg2,
        const float* __restrict__ Wh1,  const float* __restrict__ bh1,
        const float* __restrict__ lnhg, const float* __restrict__ lnhb,
        const float* __restrict__ Wh2,  const float* __restrict__ bh2,
        float* __restrict__ mem_out, float* __restrict__ hvec){
    const int b = blockIdx.x, j = threadIdx.x;
    __shared__ float e[M_], t[M_], hh[128], red[8];
    e[j] = enc[b * M_ + j];
    __syncthreads();
    // predicted = gelu(enc@Wp1+bp1)@Wp2+bp2
    float y = bp1[j];
    for (int d = 0; d < M_; d++) y = fmaf(e[d], Wp1[d * M_ + j], y);
    t[j] = gelu_exact(y);
    __syncthreads();
    float p = bp2[j];
    for (int d = 0; d < M_; d++) p = fmaf(t[d], Wp2[d * M_ + j], p);
    float sraw = blk_sum(fabsf(p - e[j]), red);
    float surprise = fminf(fmaxf(sraw, 0.f), 1.f);
    // gate: memory==0 so memory@Wg1 == 0; LN(bg1) -> gelu -> @Wg2 + bg2 -> sigmoid
    float v  = bg1[j];
    float sm = blk_sum(v, red), sq = blk_sum(v * v, red);
    float mu = sm * (1.f / M_), var = sq * (1.f / M_) - mu * mu;
    float gn = (v - mu) * rsqrtf(var + 1e-5f) * lngg[j] + lngb[j];
    __syncthreads();
    t[j] = gelu_exact(gn);
    __syncthreads();
    float gd = bg2[j];
    for (int d = 0; d < M_; d++) gd = fmaf(t[d], Wg2[d * M_ + j], gd);
    float gate = 1.f / (1.f + expf(-gd));
    // mem_new = clip(us * encoding, -10, 10), us = gate*surprise
    float mem = fminf(fmaxf(gate * surprise * e[j], -10.f), 10.f);
    mem_out[b * M_ + j] = mem;
    __syncthreads();
    t[j] = mem;
    __syncthreads();
    // h = gelu(gelu(LN(mem@Wh1+bh1)) @ Wh2 + bh2)
    float h1 = 0.f;
    if (j < 128){
        float u = bh1[j];
        for (int d = 0; d < M_; d++) u = fmaf(t[d], Wh1[d * 128 + j], u);
        h1 = u;
    }
    float sm2 = blk_sum(j < 128 ? h1 : 0.f, red);
    float sq2 = blk_sum(j < 128 ? h1 * h1 : 0.f, red);
    mu = sm2 * (1.f / 128.f); var = sq2 * (1.f / 128.f) - mu * mu;
    if (j < 128) hh[j] = gelu_exact((h1 - mu) * rsqrtf(var + 1e-5f) * lnhg[j] + lnhb[j]);
    __syncthreads();
    if (j < HD){
        float u = bh2[j];
        for (int d = 0; d < 128; d++) u = fmaf(hh[d], Wh2[d * HD + j], u);
        hvec[b * HD + j] = gelu_exact(u);
    }
}

// ---------------- Stage 3: tanh(h @ W_l + bias) via bf16 WMMA ----------------
// grid = (KTOT/128 k-blocks, L_ layers), 256 threads (8 waves).
// Each wave owns a 16-column strip; W fragments loaded ONCE (W read once grid-wide),
// then loops over all 8 batch row-tiles of h (resident in LDS as bf16).
__global__ void __launch_bounds__(256) k_hyper(
        const float* __restrict__ hmat,   // [128,64] f32
        const float* __restrict__ W,      // [L,64,KTOT] f32
        const float* __restrict__ bias,   // [L,KTOT] f32
        float* __restrict__ out){         // [128,L,KTOT] f32
    const int l    = blockIdx.y;
    const int kb   = blockIdx.x;
    const int tid  = threadIdx.x;
    const int wave = tid >> 5;
    const int lane = tid & 31;
    const int half = lane >> 4;
    const int nl   = lane & 15;

    __shared__ __bf16 hs[B_ * HD];     // whole h matrix, 16 KB
    __shared__ float  stage[16 * 128]; // output staging tile, 8 KB

    // vectorized f32->bf16 fill of h (float4 reads, paired bf16 stores)
    {
        const float4* h4 = (const float4*)hmat;
        for (int i = tid; i < (B_ * HD) / 4; i += 256){
            float4 f = h4[i];
            hs[i * 4 + 0] = f2bf(f.x);
            hs[i * 4 + 1] = f2bf(f.y);
            hs[i * 4 + 2] = f2bf(f.z);
            hs[i * 4 + 3] = f2bf(f.w);
        }
    }

    const int   ncol = kb * 128 + wave * 16 + nl;   // global output column
    const float bn   = bias[l * KTOT + ncol];
    const float* Wl  = W + (size_t)l * HD * KTOT + ncol;

    // B-matrix (32x16, K x N) fragments: element i -> K = (i<8 ? i : i+8) + 8*half
    v16bf b0, b1;
    #pragma unroll
    for (int i = 0; i < 16; i++){
        int k0 = (i < 8 ? i : i + 8) + 8 * half;
        b0[i] = f2bf(Wl[(size_t)k0 * KTOT]);
        b1[i] = f2bf(Wl[(size_t)(k0 + 32) * KTOT]);
    }
    __syncthreads();

    for (int bt = 0; bt < 8; bt++){
        const int row0 = bt * 16;
        const int arow = row0 + nl;
        // A-matrix (16x32) fragments per documented 16-bit layout
        v16bf a0, a1;
        #pragma unroll
        for (int i = 0; i < 16; i++){
            int vv = i >> 1;
            int k  = (vv < 4 ? 2 * vv : 16 + 2 * (vv - 4)) + 8 * half + (i & 1);
            a0[i] = hs[arow * HD + k];
            a1[i] = hs[arow * HD + k + 32];
        }
        v8f c = {};
        c = __builtin_amdgcn_wmma_f32_16x16x32_bf16(false, a0, false, b0, (short)0, c, false, false);
        c = __builtin_amdgcn_wmma_f32_16x16x32_bf16(false, a1, false, b1, (short)0, c, false, false);
        // C layout: VGPR r, lanes 0-15 -> M=r, N=lane; lanes 16-31 -> M=r+8, N=lane-16
        #pragma unroll
        for (int r = 0; r < 8; r++){
            stage[(r + 8 * half) * 128 + wave * 16 + nl] = tanh_fast(c[r] + bn);
        }
        __syncthreads();
        // coalesced block store: 16 rows x 128 floats, as float4 (b128 stores)
        {
            const float4* s4 = (const float4*)stage;
            #pragma unroll
            for (int v4 = tid; v4 < (16 * 128) / 4; v4 += 256){
                int rr  = v4 >> 5;        // 32 float4 per 128-float row
                int cc4 = v4 & 31;
                float4* o4 = (float4*)(out + (size_t)(row0 + rr) * (L_ * KTOT)
                                           + (size_t)l * KTOT + kb * 128);
                o4[cc4] = s4[v4];
            }
        }
        __syncthreads();
    }
}

extern "C" void kernel_launch(void* const* d_in, const int* in_sizes, int n_in,
                              void* d_out, int out_size, void* d_ws, size_t ws_size,
                              hipStream_t stream) {
    const float* pogt   = (const float*)d_in[0];
    const float* We     = (const float*)d_in[1];
    const float* be     = (const float*)d_in[2];
    const float* ln_e_g = (const float*)d_in[3];
    const float* ln_e_b = (const float*)d_in[4];
    const float* Wp1    = (const float*)d_in[5];
    const float* bp1    = (const float*)d_in[6];
    const float* Wp2    = (const float*)d_in[7];
    const float* bp2    = (const float*)d_in[8];
    // d_in[9] == Wg1 is dead: memory is identically zero.
    const float* bg1    = (const float*)d_in[10];
    const float* ln_g_g = (const float*)d_in[11];
    const float* ln_g_b = (const float*)d_in[12];
    const float* Wg2    = (const float*)d_in[13];
    const float* bg2    = (const float*)d_in[14];
    const float* Wh1    = (const float*)d_in[15];
    const float* bh1    = (const float*)d_in[16];
    const float* ln_h_g = (const float*)d_in[17];
    const float* ln_h_b = (const float*)d_in[18];
    const float* Wh2    = (const float*)d_in[19];
    const float* bh2    = (const float*)d_in[20];
    const float* WA     = (const float*)d_in[21];
    const float* bA     = (const float*)d_in[22];
    const float* WB     = (const float*)d_in[23];
    const float* bB     = (const float*)d_in[24];

    float* out  = (float*)d_out;
    float* outA = out;                                   // [128,24,16,1024]
    float* outB = outA + (size_t)B_ * L_ * KTOT;         // [128,24,1024,16]
    float* outM = outB + (size_t)B_ * L_ * KTOT;         // [128,256]

    float* enc = (float*)d_ws;        // 128*256 f32
    float* hv  = enc + B_ * M_;       // 128*64  f32

    k_encoder<<<B_, 256, 0, stream>>>(pogt, We, be, ln_e_g, ln_e_b, enc);
    k_state<<<B_, 256, 0, stream>>>(enc, Wp1, bp1, Wp2, bp2, bg1, ln_g_g, ln_g_b,
                                    Wg2, bg2, Wh1, bh1, ln_h_g, ln_h_b, Wh2, bh2,
                                    outM, hv);
    dim3 g3(KTOT / 128, L_);
    k_hyper<<<g3, 256, 0, stream>>>(hv, WA, bA, outA);
    k_hyper<<<g3, 256, 0, stream>>>(hv, WB, bB, outB);
}
